// image_aggregation_7533372638008
// MI455X (gfx1250) — matrix operand order repaired and checked by reading, not compiled, not run on hardware
//
#include <hip/hip_runtime.h>
#include <math.h>

// ---------------------------------------------------------------------------
// image_aggregation for MI455X (gfx1250, wave32, WMMA + async global->LDS)
//   B=32, C=1536, H=W=24 (N=576), HID=512, D=128, K=64
// Pipeline:
//   [cvt]  x,w1c,w1s,w2c,w2s  f32 -> bf16 (workspace), vectorized
//   [gemm] hc = relu(w1c·x + b1c)   M=512 K=1536 N=576   (bf16 out)
//   [gemm] hs = relu(w1s·x + b1s)   M=512 K=1536 N=576   (bf16 out)
//   [gemm] f  = w2c·hc + b2c        M=128 K=512  N=576   (bf16 out)
//   [gemm] s  = w2s·hs + b2s        M=64  K=512  N=576   (f32 out)
//   [sink] log-domain sinkhorn(3) + p = exp(lp/T)+exp(lp) (bf16 out)
//   [agg ] agg = f·p^T (128x64, k=576) fused with L2 normalize -> d_out f32
// GEMMs: v_wmma_f32_16x16x32_bf16; A/K-major tiles staged with
// global_load_async_to_lds_b128 (ASYNCcnt), B transposed via b128+ds_store_b16.
// ---------------------------------------------------------------------------

typedef __attribute__((ext_vector_type(16))) __bf16        v16bf;
typedef __attribute__((ext_vector_type(8)))  float         v8f;
typedef __attribute__((ext_vector_type(8)))  unsigned int  v8u;

#define BATCH 32
#define CIN   1536
#define NPIX  576
#define HIDN  512
#define DDIM  128
#define KCLU  64

__device__ __forceinline__ unsigned short f32_to_bf16(float f) {
    unsigned int u = __builtin_bit_cast(unsigned int, f);
    unsigned int r = u + 0x7FFFu + ((u >> 16) & 1u);   // round-to-nearest-even
    return (unsigned short)(r >> 16);
}

__device__ __forceinline__ v8f wmma_bf16(v16bf a, v16bf b, v8f c) {
    return __builtin_amdgcn_wmma_f32_16x16x32_bf16(
        /*neg_a=*/false, a, /*neg_b=*/false, b,
        /*c_mod=*/(short)0, c, /*reuse_a=*/false, /*reuse_b=*/false);
}

// Per-lane async copy: 16 bytes global -> LDS, tracked by ASYNCcnt.
__device__ __forceinline__ void async_copy_b128(unsigned lds_off, const void* g) {
    asm volatile("global_load_async_to_lds_b128 %0, %1, off"
                 :: "v"(lds_off), "v"(g) : "memory");
}
__device__ __forceinline__ void wait_async0() {
    asm volatile("s_wait_asynccnt 0x0" ::: "memory");
}
__device__ __forceinline__ unsigned lds_addr(const void* p) {
    return (unsigned)(size_t)p;  // low 32 bits of flat LDS addr == DS offset
}

// A fragment (16x32 bf16, ISA 7.12.2): lane L -> row M=L&15, half=L>>4.
// VGPR v<4 holds K = 8*half + 2v..2v+1 ; v>=4 holds K = 16 + 8*half + 2(v-4)..
// AsU: row-major tile, 16 dwords (32 bf16) per row.
__device__ __forceinline__ v16bf load_a_frag(const unsigned int* AsU,
                                             int row, int half) {
    v8u t;
    const unsigned int* rp = AsU + row * 16;
#pragma unroll
    for (int v = 0; v < 8; ++v) {
        int idx = 4 * half + v + ((v >= 4) ? 4 : 0);
        t[v] = rp[idx];
    }
    return __builtin_bit_cast(v16bf, t);
}

// B fragment (32x16 bf16): lane L -> col N=L&15, half=L>>4, K order matches A (8*half.., 16+8*half..).
// rowU: LDS row for that column with K contiguous (16 dwords).
__device__ __forceinline__ v16bf load_b_frag(const unsigned int* rowU, int half) {
    v8u t;
#pragma unroll
    for (int v = 0; v < 8; ++v) t[v] = rowU[4 * half + v + ((v >= 4) ? 4 : 0)];
    return __builtin_bit_cast(v16bf, t);
}

// ---------------------------------------------------------------------------
__global__ void cvt_bf16_kernel(const float* __restrict__ in,
                                unsigned short* __restrict__ out, int n) {
    int i = (blockIdx.x * blockDim.x + threadIdx.x) * 4;
    if (i + 3 < n) {
        float4 f = *(const float4*)(in + i);
        unsigned int lo = (unsigned)f32_to_bf16(f.x) |
                          ((unsigned)f32_to_bf16(f.y) << 16);
        unsigned int hi = (unsigned)f32_to_bf16(f.z) |
                          ((unsigned)f32_to_bf16(f.w) << 16);
        uint2 o; o.x = lo; o.y = hi;
        *(uint2*)(out + i) = o;
    } else {
        for (; i < n; ++i) out[i] = f32_to_bf16(in[i]);
    }
}

// ---------------------------------------------------------------------------
// Generic GEMM: OUT[b] = op(W (MxK) · IN[b] (KxN) + bias), N=576 here.
// Block: 256 threads (8 waves), 64(M) x 64(N) tile, K stepped by 32.
// A tile (64x32, K-major rows): one async b128 per lane, fixed LDS dst.
// B tile (32x64): one b128 per lane + 8 scattered ds_store_b16 (transpose).
// Wave w: m-tile (w&3)*16, n-tiles (w>>2)*32 and +16 -> 2 accumulators.
// ---------------------------------------------------------------------------
__global__ __launch_bounds__(256)
void gemm_wmma_kernel(const unsigned short* __restrict__ W16,
                      const float* __restrict__ bias,
                      const unsigned short* __restrict__ IN16,
                      unsigned short* __restrict__ OUT16,
                      float* __restrict__ OUTF,
                      int M, int K, int N, int do_relu, int out_bf16) {
    __shared__ unsigned int   AsU[64 * 16];     // 64 rows x 32 bf16
    __shared__ unsigned short BsT[64 * 32];     // [n][k]: K contiguous per col

    const int tid  = threadIdx.x;
    const int wv   = tid >> 5;
    const int lane = tid & 31;
    const int half = lane >> 4;
    const int l16  = lane & 15;

    const int nBase = blockIdx.x * 64;
    const int mBase = blockIdx.y * 64;
    const int bIdx  = blockIdx.z;

    const int mTile = (wv & 3) * 16;
    const int nW    = (wv >> 2) * 32;

    // ---- hoisted per-thread staging pointers (stride-bumped in the loop) ----
    const int aRow = tid >> 2, aQuad = tid & 3;        // A: 64 rows x 4 quads
    const unsigned short* aSrc = W16 + (size_t)(mBase + aRow) * K + 8 * aQuad;
    const unsigned ldsA = lds_addr(&AsU[aRow * 16 + aQuad * 4]);

    const int bK = tid >> 3, bN = (tid & 7) * 8;       // B: 32 k-rows x 8 octs
    const unsigned short* bSrc =
        IN16 + (size_t)bIdx * K * N + (size_t)bK * N + nBase + bN;
    unsigned short* bDst = &BsT[bN * 32 + bK];         // stride 32 u16 per n

    v8f acc0 = {};
    v8f acc1 = {};

    const int numK = K >> 5;
    for (int kc = 0; kc < numK; ++kc) {
        __syncthreads();                       // previous tile consumed
        async_copy_b128(ldsA, aSrc);           // A: global -> LDS direct
        uint4 bq = *(const uint4*)bSrc;        // B: 8 bf16 along n at row bK
        unsigned short e[8] = {
            (unsigned short)(bq.x & 0xFFFFu), (unsigned short)(bq.x >> 16),
            (unsigned short)(bq.y & 0xFFFFu), (unsigned short)(bq.y >> 16),
            (unsigned short)(bq.z & 0xFFFFu), (unsigned short)(bq.z >> 16),
            (unsigned short)(bq.w & 0xFFFFu), (unsigned short)(bq.w >> 16)};
#pragma unroll
        for (int j = 0; j < 8; ++j) bDst[j * 32] = e[j];   // transpose scatter
        aSrc += 32;
        bSrc += (size_t)32 * N;
        wait_async0();
        __syncthreads();

        v16bf a  = load_a_frag(AsU, mTile + l16, half);
        v16bf b0 = load_b_frag((const unsigned int*)&BsT[(nW + l16) * 32], half);
        v16bf b1 = load_b_frag((const unsigned int*)&BsT[(nW + 16 + l16) * 32], half);
        acc0 = wmma_bf16(a, b0, acc0);
        acc1 = wmma_bf16(a, b1, acc1);
    }

    // epilogue: C/D layout -> VGPR r: m = base + r + 8*half, n = base + (lane&15)
    const int mG = mBase + mTile + 8 * half;
    const int nG = nBase + nW + l16;
    const size_t oBase = ((size_t)bIdx * M + mG) * N + nG;
#pragma unroll
    for (int r = 0; r < 8; ++r) {
        float bv = bias[mG + r];
        float v0 = acc0[r] + bv;
        float v1 = acc1[r] + bv;
        if (do_relu) { v0 = fmaxf(v0, 0.0f); v1 = fmaxf(v1, 0.0f); }
        size_t o = oBase + (size_t)r * N;
        if (out_bf16) {
            OUT16[o]      = f32_to_bf16(v0);
            OUT16[o + 16] = f32_to_bf16(v1);
        } else {
            OUTF[o]      = v0;
            OUTF[o + 16] = v1;
        }
    }
}

// ---------------------------------------------------------------------------
// Sinkhorn (log-domain, 3 iters) per batch + p = exp(lp/T)+exp(lp) -> bf16
// One 256-thread workgroup per batch. S: (64 x 576) f32 in global.
// ---------------------------------------------------------------------------
__global__ __launch_bounds__(256)
void sinkhorn_kernel(const float* __restrict__ Sg,
                     unsigned short* __restrict__ P16,
                     const float* __restrict__ temperature) {
    __shared__ float u[KCLU];
    __shared__ float v[NPIX];

    const int b    = blockIdx.x;
    const int tid  = threadIdx.x;
    const int wv   = tid >> 5;
    const int lane = tid & 31;
    const float*  S = Sg + (size_t)b * KCLU * NPIX;
    const float norm = -logf((float)NPIX);

    for (int i = tid; i < NPIX; i += 256) v[i] = 0.0f;
    for (int i = tid; i < KCLU; i += 256) u[i] = 0.0f;
    __syncthreads();

    for (int it = 0; it < 3; ++it) {
        // u[m] = norm - LSE_n(S[m,n] + v[n]) ; one wave per row group
        for (int r = wv; r < KCLU; r += 8) {
            const float* row = S + (size_t)r * NPIX;
            float mx = -3.4e38f;
            for (int n = lane; n < NPIX; n += 32) mx = fmaxf(mx, row[n] + v[n]);
#pragma unroll
            for (int off = 16; off > 0; off >>= 1)
                mx = fmaxf(mx, __shfl_xor(mx, off, 32));
            float se = 0.0f;
            for (int n = lane; n < NPIX; n += 32) se += expf(row[n] + v[n] - mx);
#pragma unroll
            for (int off = 16; off > 0; off >>= 1)
                se += __shfl_xor(se, off, 32);
            if (lane == 0) u[r] = norm - (mx + logf(se));
        }
        __syncthreads();
        // v[n] = norm - LSE_m(S[m,n] + u[m]) ; one thread per column
        for (int n = tid; n < NPIX; n += 256) {
            float mx = -3.4e38f;
            for (int m = 0; m < KCLU; ++m)
                mx = fmaxf(mx, S[(size_t)m * NPIX + n] + u[m]);
            float se = 0.0f;
            for (int m = 0; m < KCLU; ++m)
                se += expf(S[(size_t)m * NPIX + n] + u[m] - mx);
            v[n] = norm - (mx + logf(se));
        }
        __syncthreads();
    }

    const float invT = 1.0f / temperature[0];
    for (int idx = tid; idx < KCLU * NPIX; idx += 256) {
        int m = idx / NPIX, n = idx - m * NPIX;
        float lp = S[idx] + u[m] + v[n] - norm;
        float p  = expf(lp * invT) + expf(lp);
        P16[(size_t)b * KCLU * NPIX + idx] = f32_to_bf16(p);
    }
}

// ---------------------------------------------------------------------------
// agg[b] = f[b] (128x576) · p[b]^T (576x64), fused L2 normalize over 8192
// One 256-thread workgroup per batch. 8 waves x (2x2 16x16 tiles).
// Both f and p are K-major over the contraction dim -> pure async staging.
// ---------------------------------------------------------------------------
__global__ __launch_bounds__(256)
void agg_norm_kernel(const unsigned short* __restrict__ F16,
                     const unsigned short* __restrict__ P16,
                     float* __restrict__ out) {
    __shared__ unsigned int AsU[128 * 16];  // f tile: 128 rows x 32 k
    __shared__ unsigned int BsU[64 * 16];   // p tile: 64 cols x 32 k (k contig)
    __shared__ float red[256];
    __shared__ float normsh;

    const int b    = blockIdx.x;
    const int tid  = threadIdx.x;
    const int wv   = tid >> 5;
    const int lane = tid & 31;
    const int half = lane >> 4;
    const int l16  = lane & 15;

    const int mW = (wv & 3) * 32;   // two m tiles: mW, mW+16
    const int nW = (wv >> 2) * 32;  // two col tiles: nW, nW+16

    // f tile: 512 b128 -> 2 per lane; p tile: 256 b128 -> 1 per lane
    const int row = tid >> 2, quad = tid & 3;
    const unsigned short* fSrc0 =
        F16 + (size_t)b * DDIM * NPIX + (size_t)row * NPIX + 8 * quad;
    const unsigned short* fSrc1 = fSrc0 + (size_t)64 * NPIX;
    const unsigned short* pSrc =
        P16 + (size_t)b * KCLU * NPIX + (size_t)row * NPIX + 8 * quad;
    const unsigned ldsF0 = lds_addr(&AsU[row * 16 + quad * 4]);
    const unsigned ldsF1 = lds_addr(&AsU[(row + 64) * 16 + quad * 4]);
    const unsigned ldsP  = lds_addr(&BsU[row * 16 + quad * 4]);  // row<64

    v8f a00 = {}, a01 = {}, a10 = {}, a11 = {};

    for (int kc = 0; kc < NPIX / 32; ++kc) {
        __syncthreads();
        async_copy_b128(ldsF0, fSrc0);
        async_copy_b128(ldsF1, fSrc1);
        if (row < KCLU) async_copy_b128(ldsP, pSrc);
        fSrc0 += 32; fSrc1 += 32; pSrc += 32;
        wait_async0();
        __syncthreads();

        v16bf a0 = load_a_frag(AsU, mW + l16, half);
        v16bf a1 = load_a_frag(AsU, mW + 16 + l16, half);
        v16bf b0 = load_b_frag(&BsU[(nW + l16) * 16], half);
        v16bf b1 = load_b_frag(&BsU[(nW + 16 + l16) * 16], half);
        a00 = wmma_bf16(a0, b0, a00);
        a01 = wmma_bf16(a0, b1, a01);
        a10 = wmma_bf16(a1, b0, a10);
        a11 = wmma_bf16(a1, b1, a11);
    }

    // sum of squares over all 8192 elements of this batch row
    float ss = 0.0f;
#pragma unroll
    for (int r = 0; r < 8; ++r) {
        ss += a00[r] * a00[r] + a01[r] * a01[r]
            + a10[r] * a10[r] + a11[r] * a11[r];
    }
    red[tid] = ss;
    __syncthreads();
    for (int s = 128; s > 0; s >>= 1) {
        if (tid < s) red[tid] += red[tid + s];
        __syncthreads();
    }
    if (tid == 0) normsh = fmaxf(sqrtf(red[0]), 1e-12f);
    __syncthreads();
    const float inv = 1.0f / normsh;

    float* ob = out + (size_t)b * DDIM * KCLU;
    const int m0 = mW + 8 * half;
    const int j0 = nW + l16;
#pragma unroll
    for (int r = 0; r < 8; ++r) {
        ob[(m0 + r) * KCLU + j0]            = a00[r] * inv;
        ob[(m0 + r) * KCLU + j0 + 16]       = a01[r] * inv;
        ob[(m0 + 16 + r) * KCLU + j0]       = a10[r] * inv;
        ob[(m0 + 16 + r) * KCLU + j0 + 16]  = a11[r] * inv;
    }
}

// ---------------------------------------------------------------------------
extern "C" void kernel_launch(void* const* d_in, const int* in_sizes, int n_in,
                              void* d_out, int out_size, void* d_ws, size_t ws_size,
                              hipStream_t stream) {
    const float* x    = (const float*)d_in[0];
    const float* w1c  = (const float*)d_in[1];
    const float* b1c  = (const float*)d_in[2];
    const float* w2c  = (const float*)d_in[3];
    const float* b2c  = (const float*)d_in[4];
    const float* w1s  = (const float*)d_in[5];
    const float* b1s  = (const float*)d_in[6];
    const float* w2s  = (const float*)d_in[7];
    const float* b2s  = (const float*)d_in[8];
    const float* temp = (const float*)d_in[9];
    float* out = (float*)d_out;

    // workspace carve-up (bytes, 256-aligned)
    auto align = [](size_t v) { return (v + 255) & ~(size_t)255; };
    char* ws = (char*)d_ws;
    size_t off = 0;
    unsigned short* x16   = (unsigned short*)(ws + off); off += align((size_t)BATCH * CIN * NPIX * 2);
    unsigned short* w1c16 = (unsigned short*)(ws + off); off += align((size_t)HIDN * CIN * 2);
    unsigned short* w1s16 = (unsigned short*)(ws + off); off += align((size_t)HIDN * CIN * 2);
    unsigned short* w2c16 = (unsigned short*)(ws + off); off += align((size_t)DDIM * HIDN * 2);
    unsigned short* w2s16 = (unsigned short*)(ws + off); off += align((size_t)KCLU * HIDN * 2);
    unsigned short* hc16  = (unsigned short*)(ws + off); off += align((size_t)BATCH * HIDN * NPIX * 2);
    unsigned short* hs16  = (unsigned short*)(ws + off); off += align((size_t)BATCH * HIDN * NPIX * 2);
    unsigned short* f16   = (unsigned short*)(ws + off); off += align((size_t)BATCH * DDIM * NPIX * 2);
    float*          s32   = (float*)(ws + off);          off += align((size_t)BATCH * KCLU * NPIX * 4);
    unsigned short* p16   = (unsigned short*)(ws + off); off += align((size_t)BATCH * KCLU * NPIX * 2);

    auto cvt = [&](const float* src, unsigned short* dst, int n) {
        cvt_bf16_kernel<<<(n / 4 + 255) / 256, 256, 0, stream>>>(src, dst, n);
    };
    cvt(x,   x16,   BATCH * CIN * NPIX);
    cvt(w1c, w1c16, HIDN * CIN);
    cvt(w1s, w1s16, HIDN * CIN);
    cvt(w2c, w2c16, DDIM * HIDN);
    cvt(w2s, w2s16, KCLU * HIDN);

    dim3 blk(256);
    // layer 1 (both paths): relu(W1 · x + b1)
    gemm_wmma_kernel<<<dim3(NPIX / 64, HIDN / 64, BATCH), blk, 0, stream>>>(
        w1c16, b1c, x16, hc16, nullptr, HIDN, CIN, NPIX, /*relu=*/1, /*bf16=*/1);
    gemm_wmma_kernel<<<dim3(NPIX / 64, HIDN / 64, BATCH), blk, 0, stream>>>(
        w1s16, b1s, x16, hs16, nullptr, HIDN, CIN, NPIX, /*relu=*/1, /*bf16=*/1);
    // layer 2: f (bf16), s (f32 for sinkhorn)
    gemm_wmma_kernel<<<dim3(NPIX / 64, DDIM / 64, BATCH), blk, 0, stream>>>(
        w2c16, b2c, hc16, f16, nullptr, DDIM, HIDN, NPIX, /*relu=*/0, /*bf16=*/1);
    gemm_wmma_kernel<<<dim3(NPIX / 64, KCLU / 64, BATCH), blk, 0, stream>>>(
        w2s16, b2s, hs16, nullptr, s32, KCLU, HIDN, NPIX, /*relu=*/0, /*bf16=*/0);

    sinkhorn_kernel<<<dim3(BATCH), blk, 0, stream>>>(s32, p16, temp);
    agg_norm_kernel<<<dim3(BATCH), blk, 0, stream>>>(f16, p16, out);
    (void)in_sizes; (void)n_in; (void)out_size; (void)ws_size;
}
